// ConvolutionStack_36653250904558
// MI455X (gfx1250) — compile-verified, matrix operand order checked by reading, not run on hardware
//
#include <hip/hip_runtime.h>
#include <hip/hip_bf16.h>

#define LAYERS 20
#define CCH    128      // residual channels
#define SCH    128      // skip channels
#define KW     3        // conv kernel width
#define BB     8        // batch
#define TLEN   8192     // timesteps
#define TT     64       // time tile per block
#define NT     (TT/16)  // N-tiles per block

typedef __attribute__((ext_vector_type(16))) __bf16 v16bf;
typedef __attribute__((ext_vector_type(8)))  __bf16 v8bf;
typedef __attribute__((ext_vector_type(8)))  float  v8f;
typedef __attribute__((ext_vector_type(4)))  unsigned int u32x4;
typedef unsigned short bf16raw;   // host-safe bf16 storage

// ---------------------------------------------------------------------------
// helpers
// ---------------------------------------------------------------------------
static __device__ __forceinline__ v8f vzero8() {
  v8f v = {0.f, 0.f, 0.f, 0.f, 0.f, 0.f, 0.f, 0.f};
  return v;
}

// One lane's 16-element bf16 WMMA fragment slice from a row-major bf16 row:
// elements [kb..kb+7] and [kb+16..kb+23] (CDNA5 16-bit striping: low half-K
// in VGPR0-3, high half-K in VGPR4-7).
static __device__ __forceinline__ v16bf load_frag(const __bf16* row, int kb) {
  v8bf lo = *(const v8bf*)(row + kb);
  v8bf hi = *(const v8bf*)(row + kb + 16);
  return __builtin_shufflevector(lo, hi, 0, 1, 2, 3, 4, 5, 6, 7,
                                         8, 9, 10, 11, 12, 13, 14, 15);
}

static __device__ __forceinline__ v8f wmma_bf16(v16bf a, v16bf b, v8f c) {
  return __builtin_amdgcn_wmma_f32_16x16x32_bf16(
      false, a, false, b, (short)0, c, false, false);
}

// ---------------------------------------------------------------------------
// weight prep: fp32 -> bf16, conv taps de-interleaved to [L][K][2C][C]
// ---------------------------------------------------------------------------
__global__ void prep_weights(const float* __restrict__ Wconv,
                             const float* __restrict__ Wout,
                             const float* __restrict__ Wskip,
                             bf16raw* __restrict__ wcb,
                             bf16raw* __restrict__ wob,
                             bf16raw* __restrict__ wsb) {
  const int N1 = LAYERS * KW * 2 * CCH * CCH;   // 1,966,080
  const int N2 = LAYERS * CCH * CCH;            //   327,680
  int tid = blockIdx.x * blockDim.x + threadIdx.x;
  if (tid < N1) {
    int c = tid & 127;
    int o = (tid >> 7) & 255;
    int k = (tid >> 15) % 3;
    int i = tid / (128 * 256 * 3);
    float v = Wconv[((size_t)(i * 256 + o) * 128 + c) * 3 + k];
    ((__bf16*)wcb)[tid] = (__bf16)v;
  } else if (tid < N1 + N2) {
    int j = tid - N1;
    ((__bf16*)wob)[j] = (__bf16)Wout[j];
  } else if (tid < N1 + 2 * N2) {
    int j = tid - N1 - N2;
    ((__bf16*)wsb)[j] = (__bf16)Wskip[j];
  }
}

// x0: fp32 [B][C][T]  ->  bf16 [B][T][C]  (WMMA/async-copy friendly layout)
__global__ void prep_x(const float* __restrict__ x0, bf16raw* __restrict__ xb) {
  int tid = blockIdx.x * blockDim.x + threadIdx.x;   // (b*C + c)*T + t
  int t = tid & (TLEN - 1);
  int c = (tid >> 13) & (CCH - 1);
  int b = tid >> 20;
  ((__bf16*)xb)[((size_t)b * TLEN + t) * CCH + c] = (__bf16)x0[tid];
}

// ---------------------------------------------------------------------------
// one WaveNet layer. x activations live in bf16 [B][T][C]; tap tiles are
// contiguous 16KB global regions copied via GLOBAL_LOAD_ASYNC_TO_LDS_B128.
// block = 256 threads (8 wave32) computes [all channels] x [TT time].
// LDS (48KB): tap0 (kept for residual) | tap1 (->ztile) | tap2 (->xout tile)
// ---------------------------------------------------------------------------
__global__ __launch_bounds__(256)
void wavenet_layer(const bf16raw* __restrict__ xin_,  // [B][T][C] bf16
                   bf16raw* __restrict__ xnext_,      // [B][T][C] bf16
                   const bf16raw* __restrict__ wcb_,  // [K][2C][C] bf16
                   const bf16raw* __restrict__ wob_,  // [C][C] bf16
                   const bf16raw* __restrict__ wsb_,  // [S][C] bf16
                   const float* __restrict__ bconv,   // [2C]
                   const float* __restrict__ bout,    // [C]
                   const float* __restrict__ bskip,   // [S]
                   float* __restrict__ skipdst,       // [B][S][T] this layer
                   float* __restrict__ xfinal,        // fp32 [B][C][T] (last)
                   int dil, int last) {
  const __bf16* xin = (const __bf16*)xin_;
  const __bf16* wcb = (const __bf16*)wcb_;
  const __bf16* wob = (const __bf16*)wob_;
  const __bf16* wsb = (const __bf16*)wsb_;

  __shared__ __align__(16) __bf16 lds[KW * TT * CCH];   // 48 KB
  const int th = threadIdx.x;
  const int b  = blockIdx.y;
  const int t0 = blockIdx.x * TT;
  const unsigned ldsBase = (unsigned)(uintptr_t)(void*)lds;  // LDS byte addr

  // ---- stage the 3 shifted x tiles via async global->LDS DMA ----
  for (int k = 0; k < KW; ++k) {
    const int shift = k * dil;
    int pad = (t0 < shift) ? (shift - t0) : 0;       // causal zero rows
    if (pad > TT) pad = TT;
    const int tapByte = k * TT * CCH * 2;
    // zero-fill the padded leading rows (rare; only small-t0 blocks)
    for (int jb = th * 16; jb < pad * CCH * 2; jb += 256 * 16) {
      u32x4 z4 = {0u, 0u, 0u, 0u};
      *(u32x4*)((char*)lds + tapByte + jb) = z4;
    }
    const int valid = TT - pad;
    const char* gsrc =
        (const char*)(xin + ((size_t)b * TLEN + (t0 - shift + pad)) * CCH);
    const unsigned ldst = ldsBase + (unsigned)(tapByte + pad * CCH * 2);
    const int chunks = valid * (CCH * 2 / 16);       // 16B chunks
    for (int j = th; j < chunks; j += 256) {
      unsigned lp = ldst + (unsigned)(j * 16);
      const char* gp = gsrc + (size_t)j * 16;
      asm volatile("global_load_async_to_lds_b128 %0, %1, off"
                   :: "v"(lp), "v"(gp) : "memory");
    }
  }
  asm volatile("s_wait_asynccnt 0x0" ::: "memory");
  __syncthreads();

  const int wave  = th >> 5;
  const int lane  = th & 31;
  const int lrow  = lane & 15;
  const int lhalf = lane >> 4;

  // ---- dilated conv GEMM: wave w owns rows [16w,16w+16) (tanh) and +128 ----
  v8f accA[NT], accG[NT];
#pragma unroll
  for (int n = 0; n < NT; ++n) { accA[n] = vzero8(); accG[n] = vzero8(); }

  for (int k = 0; k < KW; ++k) {
    const __bf16* Wk   = wcb + (size_t)k * 2 * CCH * CCH;
    const __bf16* tapk = lds + k * TT * CCH;
#pragma unroll
    for (int c0 = 0; c0 < CCH; c0 += 32) {
      const int kb = c0 + lhalf * 8;
      v16bf Aa = load_frag(Wk + (16 * wave + lrow) * CCH, kb);
      v16bf Ag = load_frag(Wk + (128 + 16 * wave + lrow) * CCH, kb);
#pragma unroll
      for (int n = 0; n < NT; ++n) {
        v16bf Bf = load_frag(tapk + (n * 16 + lrow) * CCH, kb);
        accA[n] = wmma_bf16(Aa, Bf, accA[n]);
        accG[n] = wmma_bf16(Ag, Bf, accG[n]);
      }
    }
  }
  __syncthreads();   // all waves done reading tap tiles

  // ---- gated activation in-register; z -> LDS [t][c] bf16 (aliases tap1) ----
  const int orow = 16 * wave + 8 * lhalf;
  float ba[8], bg[8];
#pragma unroll
  for (int r = 0; r < 8; ++r) { ba[r] = bconv[orow + r]; bg[r] = bconv[128 + orow + r]; }
  __bf16* ztile = lds + TT * CCH;
#pragma unroll
  for (int n = 0; n < NT; ++n) {
#pragma unroll
    for (int r = 0; r < 8; ++r) {
      float a = accA[n][r] + ba[r];
      float g = accG[n][r] + bg[r];
      // tanh(a)*sigmoid(g) via two v_exp_f32 + v_rcp_f32
      float sg = __builtin_amdgcn_rcpf(1.0f + __builtin_amdgcn_exp2f(-1.442695041f * g));
      float t2 = __builtin_amdgcn_rcpf(1.0f + __builtin_amdgcn_exp2f(-2.885390082f * a));
      float z = (2.0f * t2 - 1.0f) * sg;
      accA[n][r] = z;                                   // keep z for last layer
      ztile[(n * 16 + lrow) * CCH + (orow + r)] = (__bf16)z;
    }
  }
  __syncthreads();

  // ---- skip & residual-out GEMMs over z (K = 128 channels) ----
  v8f accS[NT], accO[NT];
#pragma unroll
  for (int n = 0; n < NT; ++n) { accS[n] = vzero8(); accO[n] = vzero8(); }
#pragma unroll
  for (int c0 = 0; c0 < CCH; c0 += 32) {
    const int kb = c0 + lhalf * 8;
    v16bf As = load_frag(wsb + (16 * wave + lrow) * CCH, kb);
    v16bf Ao = load_frag(wob + (16 * wave + lrow) * CCH, kb);
#pragma unroll
    for (int n = 0; n < NT; ++n) {
      v16bf Bf = load_frag(ztile + (n * 16 + lrow) * CCH, kb);
      accS[n] = wmma_bf16(As, Bf, accS[n]);
      accO[n] = wmma_bf16(Ao, Bf, accO[n]);
    }
  }

  // ---- epilogue: skip store (fp32), residual add from LDS tap0 ----
  float bs[8];
#pragma unroll
  for (int r = 0; r < 8; ++r) bs[r] = bskip[orow + r];
  const __bf16* xres = lds;                 // tap0 = x[t0..t0+TT) in [t][c]

  if (!last) {
    float bo[8];
#pragma unroll
    for (int r = 0; r < 8; ++r) bo[r] = bout[orow + r];
    __bf16* xout = lds + 2 * TT * CCH;      // aliases tap2
#pragma unroll
    for (int n = 0; n < NT; ++n) {
      const int t = t0 + n * 16 + lrow;
#pragma unroll
      for (int r = 0; r < 8; ++r) {
        const int s = orow + r;
        skipdst[((size_t)b * SCH + s) * TLEN + t] = accS[n][r] + bs[r];
        float xv = (float)xres[(n * 16 + lrow) * CCH + s];
        xout[(n * 16 + lrow) * CCH + s] = (__bf16)(accO[n][r] + bo[r] + xv);
      }
    }
    __syncthreads();
    // async store the contiguous 16KB x tile: LDS -> bf16 [B][T][C]
    __bf16* xnext = (__bf16*)xnext_;
    char* gdst = (char*)(xnext + ((size_t)b * TLEN + t0) * CCH);
    const unsigned lsrc = ldsBase + (unsigned)(2 * TT * CCH * 2);
    for (int j = th; j < TT * CCH * 2 / 16; j += 256) {
      asm volatile("global_store_async_from_lds_b128 %0, %1, off"
                   :: "v"(gdst + (size_t)j * 16), "v"(lsrc + (unsigned)(j * 16))
                   : "memory");
    }
    // implicit S_WAIT_IDLE at S_ENDPGM drains ASYNCcnt before wave exit
  } else {
    // last layer: out = z + x, written fp32 to d_out's [B][C][T] section
#pragma unroll
    for (int n = 0; n < NT; ++n) {
      const int t = t0 + n * 16 + lrow;
#pragma unroll
      for (int r = 0; r < 8; ++r) {
        const int s = orow + r;
        skipdst[((size_t)b * SCH + s) * TLEN + t] = accS[n][r] + bs[r];
        float xv = (float)xres[(n * 16 + lrow) * CCH + s];
        xfinal[((size_t)b * CCH + s) * TLEN + t] = accA[n][r] + xv;
      }
    }
  }
}

// ---------------------------------------------------------------------------
// host orchestration
// ---------------------------------------------------------------------------
extern "C" void kernel_launch(void* const* d_in, const int* in_sizes, int n_in,
                              void* d_out, int out_size, void* d_ws, size_t ws_size,
                              hipStream_t stream) {
  const float* x0    = (const float*)d_in[0];
  const float* Wconv = (const float*)d_in[1];
  const float* bconv = (const float*)d_in[2];
  const float* Wout  = (const float*)d_in[3];
  const float* bout  = (const float*)d_in[4];
  const float* Wskip = (const float*)d_in[5];
  const float* bskip = (const float*)d_in[6];
  float* out = (float*)d_out;

  const size_t XN = (size_t)BB * CCH * TLEN;   // elements per x tensor
  // ws layout: x0b | xA | xB (bf16) | wcb | wob | wsb (bf16)   (~54 MB)
  bf16raw* x0b = (bf16raw*)d_ws;
  bf16raw* xA  = x0b + XN;
  bf16raw* xB  = xA + XN;
  bf16raw* wcb = xB + XN;
  bf16raw* wob = wcb + (size_t)LAYERS * KW * 2 * CCH * CCH;
  bf16raw* wsb = wob + (size_t)LAYERS * CCH * CCH;

  {
    const int total = LAYERS * KW * 2 * CCH * CCH + 2 * LAYERS * CCH * CCH;
    prep_weights<<<(total + 255) / 256, 256, 0, stream>>>(Wconv, Wout, Wskip,
                                                          wcb, wob, wsb);
    prep_x<<<(int)(XN / 256), 256, 0, stream>>>(x0, x0b);
  }

  static const int dil[LAYERS] = {1, 2, 4, 8, 16, 32, 64, 128, 256, 512,
                                  1, 2, 4, 8, 16, 32, 64, 128, 256, 512};
  dim3 grid(TLEN / TT, BB);
  for (int i = 0; i < LAYERS; ++i) {
    const bf16raw* xi = (i == 0) ? x0b : ((i & 1) ? xA : xB);
    bf16raw* xo = (i & 1) ? xB : xA;             // unused on last layer
    float* skipdst = out + XN + (size_t)i * BB * SCH * TLEN;
    wavenet_layer<<<grid, 256, 0, stream>>>(
        xi, xo,
        wcb + (size_t)i * KW * 2 * CCH * CCH,
        wob + (size_t)i * CCH * CCH,
        wsb + (size_t)i * CCH * CCH,
        bconv + i * 2 * CCH, bout + i * CCH, bskip + i * CCH,
        skipdst, out, dil[i], (i == LAYERS - 1) ? 1 : 0);
  }
}
